// SlidingWindowAttention_71708773974020
// MI455X (gfx1250) — matrix operand order repaired and checked by reading, not compiled, not run on hardware
//
#include <hip/hip_runtime.h>
#include <hip/hip_bf16.h>
#include <stdint.h>

// ---------------------------------------------------------------------------
// Sliding-window attention (B=2, L=2048, D=1024, H=16, Dh=64, WINDOW=128)
// bf16 WMMA (v_wmma_f32_16x16x32_bf16) everywhere, f32 accumulation/softmax.
// GEMMs: block-cooperative W staging in LDS via global_load_async_to_lds_b128
// (ASYNCcnt double buffering, K-loop unrolled by the buffer period).
// Attention: flash-style banded, P transposed through LDS.
// ---------------------------------------------------------------------------

#define BQ   2
#define LQ   2048
#define DMOD 1024
#define NH   16
#define DH   64
#define WIN  128
#define NTOK (BQ*LQ)          // 4096

typedef __attribute__((ext_vector_type(16))) __bf16 v16bf;
typedef __attribute__((ext_vector_type(8)))  __bf16 v8bf;
typedef __attribute__((ext_vector_type(8)))  float  v8f;

__device__ __forceinline__ __bf16 f2bf(float f) {
    union { float f; uint32_t u; } c; c.f = f;
    uint32_t r = c.u + 0x7FFFu + ((c.u >> 16) & 1u);   // round-to-nearest-even
    unsigned short h = (unsigned short)(r >> 16);
    return __builtin_bit_cast(__bf16, h);
}

__device__ __forceinline__ v16bf loadA(const __bf16* p0, const __bf16* p1) {
    v8bf lo = *(const v8bf*)p0;
    v8bf hi = *(const v8bf*)p1;
    v16bf r;
#pragma unroll
    for (int i = 0; i < 8; ++i) { r[i] = lo[i]; r[i + 8] = hi[i]; }
    return r;
}

__device__ __forceinline__ float rmax16(float x) {
#pragma unroll
    for (int m = 1; m <= 8; m <<= 1) x = fmaxf(x, __shfl_xor(x, m, 32));
    return x;
}
__device__ __forceinline__ float rsum16(float x) {
#pragma unroll
    for (int m = 1; m <= 8; m <<= 1) x += __shfl_xor(x, m, 32);
    return x;
}

// Async global->LDS copy, 16B per lane. lds_off = LDS byte offset (addr[31:0]
// of the generic pointer), gaddr = per-lane global address. Tracked by ASYNCcnt.
__device__ __forceinline__ void async_ld_b128(uint32_t lds_off, const void* gaddr) {
    asm volatile("global_load_async_to_lds_b128 %0, %1, off"
                 :: "v"(lds_off), "v"(gaddr) : "memory");
}

// ---------------------------------------------------------------------------
__global__ __launch_bounds__(256)
void cvt_f32_bf16(const float* __restrict__ src, __bf16* __restrict__ dst, int n) {
    int i = blockIdx.x * blockDim.x + threadIdx.x;
    int stride = gridDim.x * blockDim.x;
    for (; i < n; i += stride) dst[i] = f2bf(src[i]);
}

// ---------------------------------------------------------------------------
// GEMM: Y[n,o] = sum_d X[n,d]*W[o,d] + bias[o].  X: 4096x1024, W: 1024x1024.
// Block (8 waves) -> 128(M) x 64(N); W K-panel (64x32 bf16 = 4KB) staged in
// LDS with async loads, double buffered; each wave one 16x64 slice.
// MODE 0: store bf16 head-major [B,H,L,64]   (Q, K)
// MODE 1: store bf16 transposed [B,H,64,L]   (V)
// MODE 2: store f32 row-major   [N,1024]     (final output)
// ---------------------------------------------------------------------------
template <int MODE>
__global__ __launch_bounds__(256)
void gemm16(const __bf16* __restrict__ X, const __bf16* __restrict__ W,
            const float* __restrict__ bias, __bf16* __restrict__ Yb,
            float* __restrict__ Yf) {
    __shared__ __bf16 sW[2 * 64 * 32];               // 2 x 4KB double buffer
    int tid  = threadIdx.x;
    int lane = tid & 31;
    int half = lane >> 4;                            // 0/1
    int lm   = lane & 15;
    int wv   = tid >> 5;                             // wave in block
    int mBlk = blockIdx.x >> 4;                      // 0..31  (128 rows each)
    int nT   = (blockIdx.x & 15) * 64;               // 0..960
    int mTile = mBlk * 8 + wv;                       // 0..255

    // staging map: thread t copies 16B: W row (nT + t/4), 8-elem segment t%4
    int srow = tid >> 2, sseg = tid & 3;
    const __bf16* wsrc = W + (size_t)(nT + srow) * DMOD + sseg * 8;
    uint32_t ldsbase = (uint32_t)(uintptr_t)(&sW[0]);
    uint32_t ldsoff  = ldsbase + (uint32_t)(srow * 64 + sseg * 16);

    const __bf16* arow = X + (size_t)(mTile * 16 + lm) * DMOD + half * 16;

    // prefetch K-step 0 into buffer 0
    async_ld_b128(ldsoff, wsrc);

    v8f acc[4] = {};
#pragma unroll 1
    for (int ks = 0; ks < 32; ks += 2) {             // one double-buffer period
#pragma unroll
        for (int ph = 0; ph < 2; ++ph) {
            int k = (ks + ph) * 32;
            if (ks + ph + 1 < 32) {                  // ph==0: always true
                async_ld_b128(ldsoff + (uint32_t)((ph ^ 1) * 4096), wsrc + k + 32);
                asm volatile("s_wait_asynccnt 0x1" ::: "memory");
            } else {
                asm volatile("s_wait_asynccnt 0x0" ::: "memory");
            }
            __syncthreads();                         // panel visible to all waves

            v16bf a = loadA(arow + k, arow + k + 8);
            const __bf16* lw = &sW[ph * 2048];
#pragma unroll
            for (int j = 0; j < 4; ++j) {
                v16bf bj = *(const v16bf*)(lw + (j * 16 + lm) * 32 + half * 16);
                acc[j] = __builtin_amdgcn_wmma_f32_16x16x32_bf16(
                    false, a, false, bj, (short)0, acc[j], false, false);
            }
            __syncthreads();                         // done reading before overwrite
        }
    }

    int M0 = mTile * 16 + half * 8;
#pragma unroll
    for (int j = 0; j < 4; ++j) {
        int o = nT + j * 16 + lm;
        float bj = bias[o];
#pragma unroll
        for (int v = 0; v < 8; ++v) {
            int M = M0 + v;                          // token index
            float val = acc[j][v] + bj;
            if (MODE == 2) {
                Yf[(size_t)M * DMOD + o] = val;
            } else {
                int bb = M >> 11, t = M & (LQ - 1);
                int h = o >> 6, dh = o & 63;
                size_t idx = (MODE == 0)
                    ? (((size_t)(bb * NH + h) * LQ + t) * DH + dh)
                    : (((size_t)(bb * NH + h) * DH + dh) * LQ + t);
                Yb[idx] = f2bf(val);
            }
        }
    }
}

// ---------------------------------------------------------------------------
// Banded flash attention. Qh,Kh: [B,H,L,64] bf16; Vt: [B,H,64,L] bf16.
// ctx out: [B,L,1024] bf16. One wave = one (b,h, 16-query block).
// ---------------------------------------------------------------------------
__global__ __launch_bounds__(256)
void attn_band(const __bf16* __restrict__ Qh, const __bf16* __restrict__ Kh,
               const __bf16* __restrict__ Vt, __bf16* __restrict__ ctx) {
    __shared__ __bf16 sP[8][16 * 32];            // 1KB per wave
    int lane = threadIdx.x & 31;
    int half = lane >> 4;
    int lm   = lane & 15;
    int wv   = threadIdx.x >> 5;
    int w    = blockIdx.x * 8 + wv;              // 0..4095
    int bh   = w >> 7;                           // 0..31
    int q0   = (w & 127) << 4;
    int b    = bh >> 4, h = bh & 15;

    const __bf16* Qbase = Qh + (size_t)bh * LQ * DH;
    const __bf16* Kbase = Kh + (size_t)bh * LQ * DH;
    const __bf16* Vbase = Vt + (size_t)bh * DH * LQ;

    const __bf16* qrow = Qbase + (size_t)(q0 + lm) * DH + half * 16;
    v16bf aq0 = loadA(qrow,      qrow + 8);
    v16bf aq1 = loadA(qrow + 32, qrow + 40);

    v8f o0 = {}, o1 = {}, o2 = {}, o3 = {};
    float mrun[8], srun[8];
#pragma unroll
    for (int v = 0; v < 8; ++v) { mrun[v] = -1e30f; srun[v] = 0.f; }

    int kstart = q0 - WIN; if (kstart < 0) kstart = 0;
    int kend   = q0 + 16 + WIN; if (kend > LQ) kend = LQ;
    int kb0    = kstart & ~31;

    for (int kb = kb0; kb < kend; kb += 32) {
        // ---- S = Q @ K^T (16 x 32) ----
        v8f s0 = {}, s1 = {};
        const __bf16* krow0 = Kbase + (size_t)(kb + lm) * DH + half * 16;
        const __bf16* krow1 = krow0 + 16 * DH;
        {
            v16bf bk00 = *(const v16bf*)(krow0);
            v16bf bk10 = *(const v16bf*)(krow1);
            s0 = __builtin_amdgcn_wmma_f32_16x16x32_bf16(false, aq0, false, bk00, (short)0, s0, false, false);
            s1 = __builtin_amdgcn_wmma_f32_16x16x32_bf16(false, aq0, false, bk10, (short)0, s1, false, false);
            v16bf bk01 = *(const v16bf*)(krow0 + 32);
            v16bf bk11 = *(const v16bf*)(krow1 + 32);
            s0 = __builtin_amdgcn_wmma_f32_16x16x32_bf16(false, aq1, false, bk01, (short)0, s0, false, false);
            s1 = __builtin_amdgcn_wmma_f32_16x16x32_bf16(false, aq1, false, bk11, (short)0, s1, false, false);
        }
        // ---- mask + online softmax (rows spread over 16-lane halves) ----
#pragma unroll
        for (int v = 0; v < 8; ++v) {
            int q  = q0 + half * 8 + v;
            int n0 = kb + lm, n1 = n0 + 16;
            float x0 = ((unsigned)(q - n0 + WIN) <= 2u * WIN) ? s0[v] * 0.125f : -1e30f;
            float x1 = ((unsigned)(q - n1 + WIN) <= 2u * WIN) ? s1[v] * 0.125f : -1e30f;
            float r  = rmax16(fmaxf(x0, x1));
            float mn = fmaxf(mrun[v], r);
            float corr = __expf(mrun[v] - mn);
            mrun[v] = mn;
            float p0 = __expf(x0 - mn), p1 = __expf(x1 - mn);
            srun[v] = srun[v] * corr + rsum16(p0 + p1);
            o0[v] *= corr; o1[v] *= corr; o2[v] *= corr; o3[v] *= corr;
            int row = half * 8 + v;
            sP[wv][row * 32 + lm]      = f2bf(p0);
            sP[wv][row * 32 + 16 + lm] = f2bf(p1);
        }
        asm volatile("s_wait_dscnt 0x0" ::: "memory");
        // ---- A-fragment of P from LDS (16 x 32) ----
        const __bf16* prow = &sP[wv][lm * 32 + half * 16];
        v16bf ap = loadA(prow, prow + 8);
        // ---- O += P @ V  (V transposed: B-frag contiguous) ----
        const __bf16* vr0 = Vbase + (size_t)( 0 + lm) * LQ + kb + half * 16;
        const __bf16* vr1 = Vbase + (size_t)(16 + lm) * LQ + kb + half * 16;
        const __bf16* vr2 = Vbase + (size_t)(32 + lm) * LQ + kb + half * 16;
        const __bf16* vr3 = Vbase + (size_t)(48 + lm) * LQ + kb + half * 16;
        v16bf bv0 = *(const v16bf*)vr0;
        v16bf bv1 = *(const v16bf*)vr1;
        v16bf bv2 = *(const v16bf*)vr2;
        v16bf bv3 = *(const v16bf*)vr3;
        o0 = __builtin_amdgcn_wmma_f32_16x16x32_bf16(false, ap, false, bv0, (short)0, o0, false, false);
        o1 = __builtin_amdgcn_wmma_f32_16x16x32_bf16(false, ap, false, bv1, (short)0, o1, false, false);
        o2 = __builtin_amdgcn_wmma_f32_16x16x32_bf16(false, ap, false, bv2, (short)0, o2, false, false);
        o3 = __builtin_amdgcn_wmma_f32_16x16x32_bf16(false, ap, false, bv3, (short)0, o3, false, false);
    }

    v8f oacc[4] = {o0, o1, o2, o3};
#pragma unroll
    for (int j = 0; j < 4; ++j) {
#pragma unroll
        for (int v = 0; v < 8; ++v) {
            int q = q0 + half * 8 + v;
            float val = oacc[j][v] / srun[v];
            ctx[((size_t)(b * LQ + q)) * DMOD + h * DH + j * 16 + lm] = f2bf(val);
        }
    }
}

// ---------------------------------------------------------------------------
extern "C" void kernel_launch(void* const* d_in, const int* in_sizes, int n_in,
                              void* d_out, int out_size, void* d_ws, size_t ws_size,
                              hipStream_t stream) {
    const float* q  = (const float*)d_in[0];
    const float* k  = (const float*)d_in[1];
    const float* v  = (const float*)d_in[2];
    const float* Wq = (const float*)d_in[3];
    const float* bq = (const float*)d_in[4];
    const float* Wk = (const float*)d_in[5];
    const float* bk = (const float*)d_in[6];
    const float* Wv = (const float*)d_in[7];
    const float* bv = (const float*)d_in[8];
    const float* Wo = (const float*)d_in[9];
    const float* bo = (const float*)d_in[10];
    float* out = (float*)d_out;

    char* ws = (char*)d_ws;
    const size_t SZ_ACT = (size_t)NTOK * DMOD * 2;   // 8 MB
    const size_t SZ_W   = (size_t)DMOD * DMOD * 2;   // 2 MB
    __bf16* qb  = (__bf16*)(ws);
    __bf16* kb_ = (__bf16*)(ws + SZ_ACT);
    __bf16* vb_ = (__bf16*)(ws + 2 * SZ_ACT);
    __bf16* wqb = (__bf16*)(ws + 3 * SZ_ACT);
    __bf16* wkb = (__bf16*)(ws + 3 * SZ_ACT + SZ_W);
    __bf16* wvb = (__bf16*)(ws + 3 * SZ_ACT + 2 * SZ_W);
    __bf16* wob = (__bf16*)(ws + 3 * SZ_ACT + 3 * SZ_W);
    __bf16* Qh  = (__bf16*)(ws + 3 * SZ_ACT + 4 * SZ_W);
    __bf16* Kh  = (__bf16*)(ws + 4 * SZ_ACT + 4 * SZ_W);
    __bf16* Vt  = (__bf16*)(ws + 5 * SZ_ACT + 4 * SZ_W);
    __bf16* ctx = (__bf16*)(ws + 6 * SZ_ACT + 4 * SZ_W);

    const int NA = NTOK * DMOD;          // 4M
    const int NW = DMOD * DMOD;          // 1M
    cvt_f32_bf16<<<512, 256, 0, stream>>>(q,  qb,  NA);
    cvt_f32_bf16<<<512, 256, 0, stream>>>(k,  kb_, NA);
    cvt_f32_bf16<<<512, 256, 0, stream>>>(v,  vb_, NA);
    cvt_f32_bf16<<<256, 256, 0, stream>>>(Wq, wqb, NW);
    cvt_f32_bf16<<<256, 256, 0, stream>>>(Wk, wkb, NW);
    cvt_f32_bf16<<<256, 256, 0, stream>>>(Wv, wvb, NW);
    cvt_f32_bf16<<<256, 256, 0, stream>>>(Wo, wob, NW);

    gemm16<0><<<512, 256, 0, stream>>>(qb,  wqb, bq, Qh, nullptr);
    gemm16<0><<<512, 256, 0, stream>>>(kb_, wkb, bk, Kh, nullptr);
    gemm16<1><<<512, 256, 0, stream>>>(vb_, wvb, bv, Vt, nullptr);

    attn_band<<<512, 256, 0, stream>>>(Qh, Kh, Vt, ctx);

    gemm16<2><<<512, 256, 0, stream>>>(ctx, wob, bo, nullptr, out);
}
